// GPTBigCodeAttention_20873541059257
// MI455X (gfx1250) — matrix operand order repaired and checked by reading, not compiled, not run on hardware
//
#include <hip/hip_runtime.h>
#include <hip/hip_bf16.h>

// ---------------------------------------------------------------------------
// GPTBigCode MQA attention for gfx1250 (MI455X), WMMA bf16 path, v3.
// B=2, S=2048, D=4096, H=32, HD=128, KV_DIM=128 (multi-query).
//
//   prep:  hidden f32 -> bf16 ; w_qkv,w_proj f32 -> bf16 transposed [N][K]
//   gemm1: qkvB = hiddenB @ w_qkv + b_qkv            (bf16 out)
//   flash: attnB = causal MQA softmax(qk^T/sqrt(d))v (bf16 out)
//   gemm3: out   = attnB @ w_proj + b_proj           (f32 out)
//
// Matrix math: v_wmma_f32_16x16x32_bf16.
// Hot-loop staging: global_load_async_to_lds_b128 (ASYNCcnt) where the copy
// is transpose-free; packed b64 LDS stores for the V transpose.
// ---------------------------------------------------------------------------

typedef __bf16 bf16_t;
typedef __attribute__((ext_vector_type(16))) __bf16 v16bf;
typedef __attribute__((ext_vector_type(8)))  __bf16 v8bf;
typedef __attribute__((ext_vector_type(4)))  __bf16 v4bf;
typedef __attribute__((ext_vector_type(8)))  float  v8f;

#define BATCH   2
#define SEQ     2048
#define DMODEL  4096
#define NHEADS  32
#define HDIM    128
#define QKV_N   (DMODEL + 2 * HDIM)   // 4352
#define MROWS   (BATCH * SEQ)         // 4096

// Async 16B copy global -> LDS (no VGPR round-trip, tracked by ASYNCcnt).
__device__ __forceinline__ void async_copy16(unsigned lds_addr,
                                             unsigned long long gaddr) {
    asm volatile("global_load_async_to_lds_b128 %0, %1, off"
                 :: "v"(lds_addr), "v"(gaddr) : "memory");
}
__device__ __forceinline__ void wait_async0() {
    asm volatile("s_wait_asynccnt 0" ::: "memory");
}
__device__ __forceinline__ unsigned lds_off(const void* p) {
    return (unsigned)(unsigned long long)p;   // low 32 bits = LDS offset
}

// Build a 16-element bf16 fragment from two 16-byte LDS chunks.
__device__ __forceinline__ v16bf make_frag(const bf16_t* p0, const bf16_t* p1) {
    const v8bf a = *reinterpret_cast<const v8bf*>(p0);
    const v8bf b = *reinterpret_cast<const v8bf*>(p1);
    v16bf r;
#pragma unroll
    for (int i = 0; i < 8; ++i) { r[i] = a[i]; r[i + 8] = b[i]; }
    return r;
}

// ---------------------------------------------------------------------------
// Prep 1: flat f32 -> bf16
// ---------------------------------------------------------------------------
__global__ __launch_bounds__(256) void f32_to_bf16(
    const float* __restrict__ in, bf16_t* __restrict__ out, int n4)
{
    const int i = blockIdx.x * 256 + threadIdx.x;
    if (i >= n4) return;
    const float4 f = reinterpret_cast<const float4*>(in)[i];
    v4bf o;
    o[0] = (bf16_t)f.x; o[1] = (bf16_t)f.y;
    o[2] = (bf16_t)f.z; o[3] = (bf16_t)f.w;
    *reinterpret_cast<v4bf*>(&out[(size_t)i * 4]) = o;
}

// ---------------------------------------------------------------------------
// Prep 2: W[K][N] f32 -> Wt[N][K] bf16 (32x32 LDS tile transpose)
// ---------------------------------------------------------------------------
__global__ __launch_bounds__(256) void transpose_f32_to_bf16(
    const float* __restrict__ W, bf16_t* __restrict__ Wt, int K, int N)
{
    __shared__ bf16_t tile[32][33];
    const int nb = blockIdx.x * 32;
    const int kb = blockIdx.y * 32;
    const int c  = threadIdx.x & 31;
    const int r0 = threadIdx.x >> 5;
#pragma unroll
    for (int i = 0; i < 4; ++i) {
        const int r = r0 + i * 8;
        tile[c][r] = (bf16_t)W[(size_t)(kb + r) * N + nb + c];
    }
    __syncthreads();
#pragma unroll
    for (int i = 0; i < 4; ++i) {
        const int n = r0 + i * 8;
        Wt[(size_t)(nb + n) * K + kb + c] = tile[n][c];
    }
}

// ---------------------------------------------------------------------------
// GEMM: C[M,N] = A[M,K] * Wt[N,K]^T + bias[N]   (bf16 in, fp32 acc)
// 256 threads (8 waves, 2x4), tile 128x128, K-step 32, double-buffered LDS.
// Staging = async b128 copies straight into LDS.
// ---------------------------------------------------------------------------
template <typename OutT>
__global__ __launch_bounds__(256) void gemm_bias_bf16(
    const bf16_t* __restrict__ A,   // [M,K] bf16
    const bf16_t* __restrict__ Wt,  // [N,K] bf16 (W pre-transposed)
    const float* __restrict__ bias,
    OutT* __restrict__ C, int M, int N, int K)
{
    constexpr int TM = 128, TN = 128, TK = 32;
    constexpr int LD = TK + 8;   // 40 bf16 = 80B rows (16B aligned)

    __shared__ __align__(16) bf16_t As[2][TM * LD];
    __shared__ __align__(16) bf16_t Bs[2][TN * LD];

    const int t    = threadIdx.x;
    const int wid  = t >> 5;
    const int lane = t & 31;
    const int l16  = lane & 15;
    const int half = lane >> 4;
    const int wm   = wid >> 2;    // 0..1
    const int wn   = wid & 3;     // 0..3

    const int m_blk = blockIdx.y * TM;
    const int n_blk = blockIdx.x * TN;

    // Staging: 512x 16B chunks per matrix tile, 2 per thread.
    const int c0 = t * 2;
    const int r0 = c0 >> 2,       o0 = (c0 & 3) * 8;
    const int r1 = (c0 + 1) >> 2, o1 = ((c0 + 1) & 3) * 8;

    const bf16_t* Abase = A  + (size_t)m_blk * K;
    const bf16_t* Bbase = Wt + (size_t)n_blk * K;

    const unsigned long long gA0 = (unsigned long long)(Abase + (size_t)r0 * K + o0);
    const unsigned long long gA1 = (unsigned long long)(Abase + (size_t)r1 * K + o1);
    const unsigned long long gB0 = (unsigned long long)(Bbase + (size_t)r0 * K + o0);
    const unsigned long long gB1 = (unsigned long long)(Bbase + (size_t)r1 * K + o1);

    const unsigned lA0_0 = lds_off(&As[0][r0 * LD + o0]);
    const unsigned lA0_1 = lds_off(&As[1][r0 * LD + o0]);
    const unsigned lA1_0 = lds_off(&As[0][r1 * LD + o1]);
    const unsigned lA1_1 = lds_off(&As[1][r1 * LD + o1]);
    const unsigned lB0_0 = lds_off(&Bs[0][r0 * LD + o0]);
    const unsigned lB0_1 = lds_off(&Bs[1][r0 * LD + o0]);
    const unsigned lB1_0 = lds_off(&Bs[0][r1 * LD + o1]);
    const unsigned lB1_1 = lds_off(&Bs[1][r1 * LD + o1]);

    v8f acc[4][2] = {};

    auto stage = [&](int k0, int buf) {
        const unsigned long long koff = (unsigned long long)k0 * sizeof(bf16_t);
        async_copy16(buf ? lA0_1 : lA0_0, gA0 + koff);
        async_copy16(buf ? lA1_1 : lA1_0, gA1 + koff);
        async_copy16(buf ? lB0_1 : lB0_0, gB0 + koff);
        async_copy16(buf ? lB1_1 : lB1_0, gB1 + koff);
    };

    stage(0, 0);
    const int nk = K / TK;

    for (int kt = 0; kt < nk; ++kt) {
        const int buf = kt & 1;
        wait_async0();              // my async stores to LDS have landed
        __syncthreads();            // everyone's have landed
        if (kt + 1 < nk) stage((kt + 1) * TK, buf ^ 1);
        if (kt + 2 < nk) {
            __builtin_prefetch(&Abase[(size_t)r0 * K + (size_t)(kt + 2) * TK + o0], 0, 1);
            __builtin_prefetch(&Bbase[(size_t)r0 * K + (size_t)(kt + 2) * TK + o0], 0, 1);
        }

        v16bf af[4], bfr[2];
#pragma unroll
        for (int i = 0; i < 4; ++i) {
            // A layout: lane m=lane%16; K = 16*half .. +15 contiguous
            const bf16_t* base = &As[buf][(wm * 64 + i * 16 + l16) * LD];
            af[i] = make_frag(base + 16 * half, base + 16 * half + 8);
        }
#pragma unroll
        for (int j = 0; j < 2; ++j) {
            // B layout: lane n=lane%16; K = 16*half .. +15 contiguous
            const bf16_t* base = &Bs[buf][(wn * 32 + j * 16 + l16) * LD + 16 * half];
            bfr[j] = make_frag(base, base + 8);
        }
#pragma unroll
        for (int i = 0; i < 4; ++i)
#pragma unroll
            for (int j = 0; j < 2; ++j)
                acc[i][j] = __builtin_amdgcn_wmma_f32_16x16x32_bf16(
                    false, af[i], false, bfr[j], (short)0, acc[i][j],
                    false, false);
    }

    // Epilogue: pointer-walk rows (1 add per store instead of mul_u64 chains)
#pragma unroll
    for (int j = 0; j < 2; ++j) {
        const int col = n_blk + wn * 32 + j * 16 + l16;
        const float bv = bias[col];
        OutT* p = C + (size_t)(m_blk + wm * 64 + 8 * half) * N + col;
#pragma unroll
        for (int i = 0; i < 4; ++i) {
#pragma unroll
            for (int r = 0; r < 8; ++r) {
                *p = (OutT)(acc[i][j][r] + bv);
                p += N;
            }
            p += (size_t)8 * N;   // skip other half's rows
        }
    }
}

// ---------------------------------------------------------------------------
// Flash-attention, multi-query (single shared K/V head), bf16 in/out.
// Grid: (S/64, NHEADS, B). Block: 128 threads = 4 waves, 16 q-rows per wave.
// ---------------------------------------------------------------------------
__global__ __launch_bounds__(128) void flash_mqa_wmma(
    const bf16_t* __restrict__ qkv, bf16_t* __restrict__ out)
{
    constexpr int QT = 64, KT = 64;
    constexpr int LQ = HDIM + 8;   // 136
    constexpr int LK = HDIM + 8;
    constexpr int LV = KT + 8;     // 72
    constexpr float SCALE = 0.08838834764831845f;   // 1/sqrt(128)
    constexpr float LOG2E = 1.4426950408889634f;

    __shared__ __align__(16) bf16_t Qs[QT * LQ];
    __shared__ __align__(16) bf16_t Ks[KT * LK];      // row-major [kv][d]
    __shared__ __align__(16) bf16_t Vt[HDIM * LV];    // transposed [d][kv]
    __shared__ __align__(16) bf16_t Ps[4][16 * LV];   // per-wave P tile

    const int t    = threadIdx.x;
    const int w    = t >> 5;
    const int lane = t & 31;
    const int l16  = lane & 15;
    const int half = lane >> 4;

    const int q0 = blockIdx.x * QT;
    const int h  = blockIdx.y;
    const int b  = blockIdx.z;
    const size_t base_row = (size_t)b * SEQ;

    const int d8 = (t & 15) * 8;    // 8-elem column chunk
    const int r8 = t >> 4;          // 0..7

    // Stage Q tile once (async b128 copies)
#pragma unroll
    for (int i = 0; i < 8; ++i) {
        const int q = r8 + i * 8;
        async_copy16(lds_off(&Qs[q * LQ + d8]),
                     (unsigned long long)&qkv[(base_row + q0 + q) * QKV_N +
                                              h * HDIM + d8]);
    }

    float m_i[8], l_i[8];
    v8f o[8];
#pragma unroll
    for (int r = 0; r < 8; ++r) { m_i[r] = -1e30f; l_i[r] = 0.0f; }
#pragma unroll
    for (int dt = 0; dt < 8; ++dt) o[dt] = v8f{};

    const int kv_tiles = ((q0 + QT - 1) / KT) + 1;   // causal stop

    for (int jt = 0; jt < kv_tiles; ++jt) {
        const int k0 = jt * KT;
        __syncthreads();   // previous tile's compute finished

        // Stage K row-major: async b128 copies
#pragma unroll
        for (int i = 0; i < 8; ++i) {
            const int kv = r8 + i * 8;
            async_copy16(lds_off(&Ks[kv * LK + d8]),
                         (unsigned long long)&qkv[(base_row + k0 + kv) * QKV_N +
                                                  DMODEL + d8]);
        }
        // Stage V transposed: 4-row quads -> packed b64 stores
#pragma unroll
        for (int i = 0; i < 2; ++i) {
            const int kv0 = ((t >> 4) + 8 * i) * 4;   // 0,4,..,60
            const bf16_t* vp =
                &qkv[(base_row + k0 + kv0) * QKV_N + DMODEL + HDIM + d8];
            const v8bf a0 = *(const v8bf*)(vp);
            const v8bf a1 = *(const v8bf*)(vp + QKV_N);
            const v8bf a2 = *(const v8bf*)(vp + 2 * QKV_N);
            const v8bf a3 = *(const v8bf*)(vp + 3 * QKV_N);
#pragma unroll
            for (int j = 0; j < 8; ++j) {
                v4bf pk;
                pk[0] = a0[j]; pk[1] = a1[j]; pk[2] = a2[j]; pk[3] = a3[j];
                *(v4bf*)&Vt[(d8 + j) * LV + kv0] = pk;
            }
        }
        wait_async0();
        __syncthreads();

        // S = Q * K^T : 16(q) x 64(kv) per wave, 4 K-steps of 32
        v8f s[4];
#pragma unroll
        for (int n = 0; n < 4; ++n) s[n] = v8f{};
#pragma unroll
        for (int kk = 0; kk < HDIM; kk += 32) {
            const bf16_t* ab = &Qs[(w * 16 + l16) * LQ + kk];
            const v16bf aq = make_frag(ab + 16 * half, ab + 16 * half + 8);
#pragma unroll
            for (int n = 0; n < 4; ++n) {
                const bf16_t* bb = &Ks[(n * 16 + l16) * LK + kk + 16 * half];
                const v16bf bk = make_frag(bb, bb + 8);
                s[n] = __builtin_amdgcn_wmma_f32_16x16x32_bf16(
                    false, aq, false, bk, (short)0, s[n], false, false);
            }
        }

        // Scale + causal mask + online softmax (row stats per 16-lane half)
#pragma unroll
        for (int r = 0; r < 8; ++r) {
            const int qg = q0 + w * 16 + r + 8 * half;
            float mx = m_i[r];
#pragma unroll
            for (int n = 0; n < 4; ++n) {
                const int kg = k0 + n * 16 + l16;
                float sv = s[n][r] * SCALE;
                sv = (kg <= qg) ? sv : -1e30f;
                s[n][r] = sv;
                mx = fmaxf(mx, sv);
            }
#pragma unroll
            for (int off = 1; off < 16; off <<= 1)
                mx = fmaxf(mx, __shfl_xor(mx, off, 32));
            const float corr = __builtin_amdgcn_exp2f((m_i[r] - mx) * LOG2E);
            m_i[r] = mx;
            float psum = 0.0f;
#pragma unroll
            for (int n = 0; n < 4; ++n) {
                const float p = __builtin_amdgcn_exp2f((s[n][r] - mx) * LOG2E);
                Ps[w][(r + 8 * half) * LV + n * 16 + l16] = (bf16_t)p;
                psum += p;
            }
#pragma unroll
            for (int off = 1; off < 16; off <<= 1)
                psum += __shfl_xor(psum, off, 32);
            l_i[r] = l_i[r] * corr + psum;
#pragma unroll
            for (int dt = 0; dt < 8; ++dt) o[dt][r] *= corr;
        }

        // O += P * V : A = P (16x64), B = Vt (kv x d), 2 K-steps of 32
#pragma unroll
        for (int kk = 0; kk < KT; kk += 32) {
            const bf16_t* ab = &Ps[w][l16 * LV + kk];
            const v16bf ap = make_frag(ab + 16 * half, ab + 16 * half + 8);
#pragma unroll
            for (int dt = 0; dt < 8; ++dt) {
                const bf16_t* bb = &Vt[(dt * 16 + l16) * LV + kk + 16 * half];
                const v16bf bv = make_frag(bb, bb + 8);
                o[dt] = __builtin_amdgcn_wmma_f32_16x16x32_bf16(
                    false, ap, false, bv, (short)0, o[dt], false, false);
            }
        }
    }

    // Normalize and write attn output (bf16), pointer-walk rows
    bf16_t* p = out + (base_row + q0 + w * 16 + 8 * half) * DMODEL +
                h * HDIM + l16;
#pragma unroll
    for (int r = 0; r < 8; ++r) {
        const float inv = 1.0f / l_i[r];
#pragma unroll
        for (int dt = 0; dt < 8; ++dt)
            p[dt * 16] = (bf16_t)(o[dt][r] * inv);
        p += DMODEL;
    }
}

// ---------------------------------------------------------------------------
// Launch: prep (convert/transpose) -> QKV GEMM -> flash MQA -> proj GEMM
// ---------------------------------------------------------------------------
extern "C" void kernel_launch(void* const* d_in, const int* in_sizes, int n_in,
                              void* d_out, int out_size, void* d_ws,
                              size_t ws_size, hipStream_t stream)
{
    const float* hidden = (const float*)d_in[0];
    const float* w_qkv  = (const float*)d_in[1];
    const float* b_qkv  = (const float*)d_in[2];
    const float* w_proj = (const float*)d_in[3];
    const float* b_proj = (const float*)d_in[4];
    float* out = (float*)d_out;

    // Workspace layout (bf16 elements)
    bf16_t* hiddenB = (bf16_t*)d_ws;                          // [4096,4096]
    bf16_t* wqkvT   = hiddenB + (size_t)MROWS * DMODEL;       // [4352,4096]
    bf16_t* wprojT  = wqkvT   + (size_t)QKV_N * DMODEL;       // [4096,4096]
    bf16_t* qkvB    = wprojT  + (size_t)DMODEL * DMODEL;      // [4096,4352]
    bf16_t* attnB   = qkvB    + (size_t)MROWS * QKV_N;        // [4096,4096]

    // Prep: convert activations, convert+transpose weights (one-time)
    {
        const int n4 = (MROWS * DMODEL) / 4;
        f32_to_bf16<<<dim3((n4 + 255) / 256), dim3(256), 0, stream>>>(
            hidden, hiddenB, n4);
        transpose_f32_to_bf16<<<dim3(QKV_N / 32, DMODEL / 32), dim3(256), 0,
                                stream>>>(w_qkv, wqkvT, DMODEL, QKV_N);
        transpose_f32_to_bf16<<<dim3(DMODEL / 32, DMODEL / 32), dim3(256), 0,
                                stream>>>(w_proj, wprojT, DMODEL, DMODEL);
    }

    // 1) qkvB = hiddenB @ w_qkv + b_qkv  (bf16 out)
    gemm_bias_bf16<bf16_t>
        <<<dim3(QKV_N / 128, MROWS / 128), dim3(256), 0, stream>>>(
            hiddenB, wqkvT, b_qkv, qkvB, MROWS, QKV_N, DMODEL);

    // 2) attnB = causal MQA attention (bf16 out)
    flash_mqa_wmma<<<dim3(SEQ / 64, NHEADS, BATCH), dim3(128), 0, stream>>>(
        qkvB, attnB);

    // 3) out = attnB @ w_proj + b_proj  (fp32 out)
    gemm_bias_bf16<float>
        <<<dim3(DMODEL / 128, MROWS / 128), dim3(256), 0, stream>>>(
            attnB, wprojT, b_proj, out, MROWS, DMODEL, DMODEL);
}
